// SelfAttention_83769042141764
// MI455X (gfx1250) — matrix operand order repaired and checked by reading, not compiled, not run on hardware
//
#include <hip/hip_runtime.h>
#include <hip/hip_bf16.h>

// ---------------------------------------------------------------------------
// Self-attention (B=4, C=512, CQ=128, N=64*64=4096) for MI455X / gfx1250.
//   1) cvt_weights : f32 -> bf16 weight copies in workspace
//   2) proj_kernel : WMMA bf16 GEMMs producing f_t (N x CQ), k_t (N x CQ),
//                    v_t (N x C) per batch (query-major layouts)
//   3) attn_kernel : fused flash-attention (online softmax), double-buffered
//                    async global->LDS staging of K/V tiles, WMMA for scores
//                    and att*V, + alpha scale + residual add, f32 output.
// ---------------------------------------------------------------------------

#define BDIM   4
#define CDIM   512
#define CQDIM  128
#define NDIM   4096

typedef __attribute__((ext_vector_type(16))) __bf16 v16bf;
typedef __attribute__((ext_vector_type(8)))  float  v8f;

// ---- CDNA5 async global->LDS path (ASYNCcnt), probed with __has_builtin ----
#if defined(__HIP_DEVICE_COMPILE__) && \
    __has_builtin(__builtin_amdgcn_global_load_async_to_lds_b128) && \
    __has_builtin(__builtin_amdgcn_s_wait_asynccnt)
#define HAVE_ASYNC_LDS 1
#else
#define HAVE_ASYNC_LDS 0
#endif

#if HAVE_ASYNC_LDS
typedef int v4i __attribute__((vector_size(16)));
typedef __attribute__((address_space(1))) v4i gv4i;
typedef __attribute__((address_space(3))) v4i lv4i;
__device__ __forceinline__ void async_copy_b128(void* lds_dst, const void* gsrc) {
    gv4i* g = (gv4i*)(v4i*)gsrc;
    lv4i* l = (lv4i*)(v4i*)lds_dst;
    __builtin_amdgcn_global_load_async_to_lds_b128(g, l, /*offset=*/0, /*cpol=*/0);
}
#endif

// ---- WMMA operand layout helpers (CDNA5 ISA 7.12.2, wave32) ----------------

// A-matrix 16x32 bf16 from row-major (M x K) memory, stride ld (elements).
__device__ __forceinline__ v16bf load_A(const __bf16* __restrict__ src, int ld) {
    const int lane = threadIdx.x & 31;
    const int m    = lane & 15;
    const int kb   = lane & 16;                 // 0 or 16
    v16bf a;
#pragma unroll
    for (int r = 0; r < 8; ++r) {
        const int k = 2 * r + kb;
        a[2 * r]     = src[m * ld + k];
        a[2 * r + 1] = src[m * ld + k + 1];
    }
    return a;
}

// B-matrix 32x16 bf16 from row-major (K x N) memory, stride ld.
__device__ __forceinline__ v16bf load_B(const __bf16* __restrict__ src, int ld) {
    const int lane = threadIdx.x & 31;
    const int n    = lane & 15;
    const int kh   = lane & 16;                 // 0 or 16
    v16bf b;
#pragma unroll
    for (int r = 0; r < 8; ++r) {
        const int k = kh + 2 * r;
        b[2 * r]     = src[k * ld + n];
        b[2 * r + 1] = src[(k + 1) * ld + n];
    }
    return b;
}

// B-matrix 32x16 bf16 where memory is (N x K) row-major (transposed source).
__device__ __forceinline__ v16bf load_BT(const __bf16* __restrict__ src, int ld) {
    const int lane = threadIdx.x & 31;
    const int n    = lane & 15;
    const int kh   = lane & 16;
    v16bf b;
#pragma unroll
    for (int r = 0; r < 8; ++r) {
        const int k = kh + 2 * r;
        b[2 * r]     = src[n * ld + k];
        b[2 * r + 1] = src[n * ld + k + 1];
    }
    return b;
}

#define WMMA_BF16(a, b, c) \
    __builtin_amdgcn_wmma_f32_16x16x32_bf16(false, (a), false, (b), (short)0, (c), false, false)

// ---------------------------------------------------------------------------
// Kernel 1: convert the three weight matrices to bf16.
// ---------------------------------------------------------------------------
__global__ __launch_bounds__(256) void cvt_weights(
    const float* __restrict__ wf, const float* __restrict__ wh,
    const float* __restrict__ wl,
    __bf16* __restrict__ wfb, __bf16* __restrict__ whb, __bf16* __restrict__ wlb)
{
    const int i = blockIdx.x * 256 + threadIdx.x;
    if (i < CQDIM * CDIM) { wfb[i] = (__bf16)wf[i]; whb[i] = (__bf16)wh[i]; }
    if (i < CDIM * CDIM)  { wlb[i] = (__bf16)wl[i]; }
}

// ---------------------------------------------------------------------------
// Kernel 2: projections. Block = 128 threads (4 waves), 64 queries per block.
// ---------------------------------------------------------------------------
__global__ __launch_bounds__(128) void proj_kernel(
    const float* __restrict__ x,
    const __bf16* __restrict__ wf, const float* __restrict__ biasf,
    const __bf16* __restrict__ wh, const float* __restrict__ biash,
    const __bf16* __restrict__ wl, const float* __restrict__ biasl,
    __bf16* __restrict__ f_t, __bf16* __restrict__ k_t, __bf16* __restrict__ v_t)
{
    __shared__ __attribute__((aligned(16))) __bf16 xT[64 * CDIM];   // 64 KB

    const int blk = blockIdx.x;
    const int b   = blk >> 6;                   // 64 query-tiles per batch
    const int i0  = (blk & 63) * 64;
    const float* xb = x + (size_t)b * CDIM * NDIM + i0;

    for (int id = threadIdx.x; id < CDIM * 64; id += 128) {
        const int c = id >> 6, ir = id & 63;
        xT[ir * CDIM + c] = (__bf16)xb[(size_t)c * NDIM + ir];
    }
    __syncthreads();

    const int wave = threadIdx.x >> 5;
    const int lane = threadIdx.x & 31;
    const int half = lane >> 4;
    const int q0   = i0 + wave * 16;
    const __bf16* Abase = xT + (wave * 16) * CDIM;

    const __bf16* wptr[6] = { wf, wh, wl, wl, wl, wl };
    const float*  bptr[6] = { biasf, biash, biasl, biasl, biasl, biasl };
    __bf16* f_b = f_t + (size_t)b * NDIM * CQDIM;
    __bf16* k_b = k_t + (size_t)b * NDIM * CQDIM;
    __bf16* v_b = v_t + (size_t)b * NDIM * CDIM;
    __bf16*       dptr[6] = { f_b, k_b, v_b, v_b, v_b, v_b };
    const int     ldo[6]  = { CQDIM, CQDIM, CDIM, CDIM, CDIM, CDIM };
    const int     ob[6]   = { 0, 0, 0, 128, 256, 384 };

#pragma unroll
    for (int ch = 0; ch < 6; ++ch) {
        v8f acc[8] = {};
        for (int ks = 0; ks < CDIM / 32; ++ks) {     // 16 K-steps
            const v16bf a = load_A(Abase + ks * 32, CDIM);
#pragma unroll
            for (int t = 0; t < 8; ++t) {
                const __bf16* wsrc = wptr[ch] + (size_t)(ob[ch] + t * 16) * CDIM + ks * 32;
                const v16bf bm = load_BT(wsrc, CDIM);
                acc[t] = WMMA_BF16(a, bm, acc[t]);
            }
        }
#pragma unroll
        for (int t = 0; t < 8; ++t) {
            const int o    = ob[ch] + t * 16 + (lane & 15);
            const float bi = bptr[ch][o];
#pragma unroll
            for (int r = 0; r < 8; ++r) {
                const int q = q0 + r + 8 * half;
                dptr[ch][(size_t)q * ldo[ch] + o] = (__bf16)(acc[t][r] + bi);
            }
        }
    }
}

// ---------------------------------------------------------------------------
// Kernel 3: fused attention + residual.
// Block = 256 threads (8 waves), 32 queries per block (2 groups x 16).
// Each wave: its group's 16x32 score tile + softmax + a 128-channel O slice.
// K/V tiles double-buffered in LDS, staged with async global->LDS copies.
// ---------------------------------------------------------------------------
__global__ __launch_bounds__(256) void attn_kernel(
    const float* __restrict__ x, const float* __restrict__ alpha,
    const __bf16* __restrict__ f_t, const __bf16* __restrict__ k_t,
    const __bf16* __restrict__ v_t, float* __restrict__ out)
{
    __shared__ __attribute__((aligned(16))) __bf16 klds[2][32 * CQDIM]; // 16 KB
    __shared__ __attribute__((aligned(16))) __bf16 vlds[2][32 * CDIM];  // 64 KB
    __shared__ __attribute__((aligned(16))) __bf16 plds[8][16 * 32];    //  8 KB

    const int blk  = blockIdx.x;
    const int b    = blk >> 7;                  // 128 query-tiles(32) per batch
    const int q0   = (blk & 127) * 32;
    const int wave = threadIdx.x >> 5;
    const int lane = threadIdx.x & 31;
    const int half = lane >> 4;
    const int qgrp = wave >> 2;                 // 0/1: which 16-query group
    const int cbase = (wave & 3) * 128;         // this wave's channel slice

    const __bf16* fb = f_t + ((size_t)b * NDIM + q0 + 16 * qgrp) * CQDIM;
    const __bf16* kb = k_t + (size_t)b * NDIM * CQDIM;
    const __bf16* vb = v_t + (size_t)b * NDIM * CDIM;

    // Preload query A-tiles: 16 queries x 128 channels -> 4 K-step tiles
    v16bf af[4];
#pragma unroll
    for (int ks = 0; ks < 4; ++ks) af[ks] = load_A(fb + ks * 32, CQDIM);

    v8f   acc[8] = {};
    float mr[8], lr[8];
#pragma unroll
    for (int r = 0; r < 8; ++r) { mr[r] = -3.0e38f; lr[r] = 0.0f; }

    const int tid = threadIdx.x;

    // ---- staging helpers (async if available, sync fallback) --------------
    auto stage = [&](int jt, int buf) {
        const uint4* ksrc = (const uint4*)(kb + (size_t)jt * 32 * CQDIM);
        uint4*       kdst = (uint4*)&klds[buf][0];
        const uint4* vsrc = (const uint4*)(vb + (size_t)jt * 32 * CDIM);
        uint4*       vdst = (uint4*)&vlds[buf][0];
#if HAVE_ASYNC_LDS
        for (int id = tid; id < (32 * CQDIM) / 8; id += 256) async_copy_b128(kdst + id, ksrc + id);
        for (int id = tid; id < (32 * CDIM) / 8;  id += 256) async_copy_b128(vdst + id, vsrc + id);
#else
        for (int id = tid; id < (32 * CQDIM) / 8; id += 256) kdst[id] = ksrc[id];
        for (int id = tid; id < (32 * CDIM) / 8;  id += 256) vdst[id] = vsrc[id];
#endif
    };
    auto stage_wait = [&]() {
#if HAVE_ASYNC_LDS
        __builtin_amdgcn_s_wait_asynccnt(0);
#endif
        __syncthreads();
    };

    const int NT = NDIM / 32;                   // 128 key tiles
    stage(0, 0);
    stage_wait();

    for (int jt = 0; jt < NT; ++jt) {
        const int cur = jt & 1;

        // kick off staging of the next tile into the other buffer
        if (jt + 1 < NT) stage(jt + 1, cur ^ 1);

        // --- scores S (16 x 32) : two 16x16 accumulators -------------------
        v8f s0 = {}, s1 = {};
#pragma unroll
        for (int ks = 0; ks < 4; ++ks) {
            const v16bf b0 = load_BT(&klds[cur][0 * CQDIM]  + ks * 32, CQDIM);
            const v16bf b1 = load_BT(&klds[cur][16 * CQDIM] + ks * 32, CQDIM);
            s0 = WMMA_BF16(af[ks], b0, s0);
            s1 = WMMA_BF16(af[ks], b1, s1);
        }

        // --- online softmax (rows live in 16-lane halves) ------------------
        float scale[8];
#pragma unroll
        for (int r = 0; r < 8; ++r) {
            float t = fmaxf(s0[r], s1[r]);
            t = fmaxf(t, __shfl_xor(t, 1, 32));
            t = fmaxf(t, __shfl_xor(t, 2, 32));
            t = fmaxf(t, __shfl_xor(t, 4, 32));
            t = fmaxf(t, __shfl_xor(t, 8, 32));
            const float mnew = fmaxf(mr[r], t);
            const float sc   = __expf(mr[r] - mnew);
            const float e0   = __expf(s0[r] - mnew);
            const float e1   = __expf(s1[r] - mnew);
            float rs = e0 + e1;
            rs += __shfl_xor(rs, 1, 32);
            rs += __shfl_xor(rs, 2, 32);
            rs += __shfl_xor(rs, 4, 32);
            rs += __shfl_xor(rs, 8, 32);
            mr[r]    = mnew;
            lr[r]    = lr[r] * sc + rs;
            scale[r] = sc;
            const int row = r + 8 * half;
            plds[wave][row * 32 + (lane & 15)]      = (__bf16)e0;
            plds[wave][row * 32 + 16 + (lane & 15)] = (__bf16)e1;
        }

        // rescale running output accumulators
#pragma unroll
        for (int t = 0; t < 8; ++t)
#pragma unroll
            for (int r = 0; r < 8; ++r) acc[t][r] *= scale[r];

        // --- O += P * V for this wave's 128-channel slice ------------------
        const v16bf ap = load_A(&plds[wave][0], 32);
#pragma unroll
        for (int t = 0; t < 8; ++t) {
            const v16bf bv = load_B(&vlds[cur][0] + cbase + t * 16, CDIM);
            acc[t] = WMMA_BF16(ap, bv, acc[t]);
        }

        // wait for async staging + keep waves in lockstep on LDS buffers
        stage_wait();
    }

    // --- epilogue: O/l * alpha + x ----------------------------------------
    const float al = alpha[0];
    float inv[8];
#pragma unroll
    for (int r = 0; r < 8; ++r) inv[r] = 1.0f / lr[r];

    const float* xb2 = x   + (size_t)b * CDIM * NDIM;
    float*       ob  = out + (size_t)b * CDIM * NDIM;
#pragma unroll
    for (int t = 0; t < 8; ++t) {
        const int c = cbase + t * 16 + (lane & 15);
#pragma unroll
        for (int r = 0; r < 8; ++r) {
            const int    i   = q0 + 16 * qgrp + r + 8 * half;
            const size_t idx = (size_t)c * NDIM + i;
            ob[idx] = acc[t][r] * inv[r] * al + xb2[idx];
        }
    }
}

// ---------------------------------------------------------------------------
// Host-side launch
// ---------------------------------------------------------------------------
extern "C" void kernel_launch(void* const* d_in, const int* in_sizes, int n_in,
                              void* d_out, int out_size, void* d_ws, size_t ws_size,
                              hipStream_t stream) {
    const float* x     = (const float*)d_in[0];
    const float* Wf_w  = (const float*)d_in[1];
    const float* Wf_b  = (const float*)d_in[2];
    const float* Wh_w  = (const float*)d_in[3];
    const float* Wh_b  = (const float*)d_in[4];
    const float* Wl_w  = (const float*)d_in[5];
    const float* Wl_b  = (const float*)d_in[6];
    const float* alpha = (const float*)d_in[7];
    float* out = (float*)d_out;

    // Workspace carving (bf16 elements); total ~24.75 MiB
    __bf16* wfb = (__bf16*)d_ws;
    __bf16* whb = wfb + (size_t)CQDIM * CDIM;
    __bf16* wlb = whb + (size_t)CQDIM * CDIM;
    __bf16* f_t = wlb + (size_t)CDIM * CDIM;
    __bf16* k_t = f_t + (size_t)BDIM * NDIM * CQDIM;
    __bf16* v_t = k_t + (size_t)BDIM * NDIM * CQDIM;

    cvt_weights<<<(CDIM * CDIM + 255) / 256, 256, 0, stream>>>(
        Wf_w, Wh_w, Wl_w, wfb, whb, wlb);

    proj_kernel<<<BDIM * (NDIM / 64), 128, 0, stream>>>(
        x, wfb, Wf_b, whb, Wh_b, wlb, Wl_b, f_t, k_t, v_t);

    attn_kernel<<<BDIM * (NDIM / 32), 256, 0, stream>>>(
        x, alpha, f_t, k_t, v_t, out);
}
